// LinformerMultiHeadAttention_1176821039315
// MI455X (gfx1250) — compile-verified, matrix operand order checked by reading, not run on hardware
//
#include <hip/hip_runtime.h>

// ---------------------------------------------------------------------------
// Linformer MHA forward for MI455X (gfx1250, wave32):
//   - all matmuls on v_wmma_f32_16x16x32_bf16 (fp32 accum)
//   - GEMM staging: global_load_async_to_lds_b128 + ASYNCcnt double buffering
//   - attention staging: Tensor Data Mover (tensor_load_to_lds + TENSORcnt)
// ---------------------------------------------------------------------------

typedef __bf16 bf_t;
typedef __attribute__((ext_vector_type(4)))  __bf16 v4bf;
typedef __attribute__((ext_vector_type(8)))  __bf16 v8bf;
typedef __attribute__((ext_vector_type(16))) __bf16 v16bf;
typedef __attribute__((ext_vector_type(8)))  float  v8f;
typedef __attribute__((ext_vector_type(4)))  unsigned u32x4;
typedef __attribute__((ext_vector_type(8)))  int      i32x8;
typedef __attribute__((ext_vector_type(4)))  int      i32x4;

#define DM    512
#define NH    8
#define DEPTH 64
#define RP    256
#define SEQ   4096
#define NB    8
#define NTOK  (NB * SEQ)   // 32768

__device__ __forceinline__ v8f wmma_bf16(v16bf a, v16bf b, v8f c) {
  // D = A(16x32) * B(32x16) + C(16x16 f32)
  return __builtin_amdgcn_wmma_f32_16x16x32_bf16(false, a, false, b,
                                                 (short)0, c, false, false);
}

// CDNA5 async copy: 16B global -> LDS, tracked by ASYNCcnt (no VGPR round-trip)
__device__ __forceinline__ void async_ld_b128(void* lds, const void* g) {
  unsigned off = (unsigned)(uintptr_t)lds;  // LDS aperture: addr[31:0] = LDS addr
  asm volatile("global_load_async_to_lds_b128 %0, %1, off"
               :: "v"(off), "v"(g) : "memory");
}
__device__ __forceinline__ void wait_async0() {
  asm volatile("s_wait_asynccnt 0x0" ::: "memory");
}

#if __has_builtin(__builtin_amdgcn_tensor_load_to_lds)
#define HAVE_TDM 1
// Tensor Data Mover: DMA a 2D tile (tile_d0 elems/row x tile_d1 rows, bf16,
// row stride stride_d0 elems) from global into LDS, appending pad_amount
// DWORDs of LDS padding every 2^(pad_interval+1) DWORDs (D# Sec 8.3/8.4).
// This toolchain exposes the 6-arg builtin:
//   (u32x4 group0, i32x8 group1, i32x4 group2, i32x4 group3, i32x8, i32 cpol)
__device__ __forceinline__ void tdm_load_2d(void* lds, const void* gaddr,
                                            unsigned tile_d0, unsigned tile_d1,
                                            unsigned stride_d0,
                                            unsigned pad_interval_code,
                                            unsigned pad_amount_code) {
  const unsigned long long ga = (unsigned long long)(uintptr_t)gaddr;
  u32x4 g0;
  g0[0] = 1u;                                         // count=1 (valid user D#)
  g0[1] = (unsigned)(uintptr_t)lds;                   // lds_addr
  g0[2] = (unsigned)(ga & 0xFFFFFFFFu);               // global_addr[31:0]
  g0[3] = (unsigned)((ga >> 32) & 0x01FFFFFFu) | (2u << 30);  // addr[56:32],type=2
  i32x8 g1;
  g1[0] = (int)((1u << 16) |                          // data_size=1 -> 2 bytes
                (1u << 20) |                          // pad_enable
                (pad_interval_code << 22) |
                (pad_amount_code << 25));
  g1[1] = (int)((tile_d0 & 0xFFFFu) << 16);           // tensor_dim0[15:0]
  g1[2] = (int)(((tile_d0 >> 16) & 0xFFFFu) |         // tensor_dim0[31:16]
                ((tile_d1 & 0xFFFFu) << 16));         // tensor_dim1[15:0]
  g1[3] = (int)(((tile_d1 >> 16) & 0xFFFFu) |         // tensor_dim1[31:16]
                ((tile_d0 & 0xFFFFu) << 16));         // tile_dim0
  g1[4] = (int)(tile_d1 & 0xFFFFu);                   // tile_dim1 (tile_dim2=0)
  g1[5] = (int)stride_d0;                             // tensor_dim0_stride[31:0]
  g1[6] = 0;                                          // stride hi / dim1_stride
  g1[7] = 0;
  const i32x4 z4 = {0, 0, 0, 0};                      // groups 2/3 unused (2D)
  const i32x8 z8 = {0, 0, 0, 0, 0, 0, 0, 0};
  __builtin_amdgcn_tensor_load_to_lds(g0, g1, z4, z4, z8, 0);
}
#endif

// Load one 16-lane bf16 fragment: per lane two contiguous 16B runs
// (K in [kbase+hi*8, +8) and [kbase+16+hi*8, +8)) -> two ds_load_b128.
__device__ __forceinline__ v16bf ld_frag(const bf_t* p) {
  v8bf lo = *(const v8bf*)(p);
  v8bf hi = *(const v8bf*)(p + 16);
  v16bf r;
#pragma unroll
  for (int i = 0; i < 8; ++i) { r[i] = lo[i]; r[i + 8] = hi[i]; }
  return r;
}

__device__ __forceinline__ unsigned pack_bf2(float a, float b) {
  bf_t ha = (bf_t)a, hb = (bf_t)b;
  unsigned short ua, ub;
  __builtin_memcpy(&ua, &ha, 2);
  __builtin_memcpy(&ub, &hb, 2);
  return (unsigned)ua | ((unsigned)ub << 16);
}

// ---------------------------------------------------------------------------
// fp32 -> bf16 converts
// ---------------------------------------------------------------------------
__global__ __launch_bounds__(256)
void cvt_copy4(const float4* __restrict__ src, v4bf* __restrict__ dst) {
  const int i = blockIdx.x * 256 + threadIdx.x;
  float4 f = src[i];
  v4bf o;
  o[0] = (bf_t)f.x; o[1] = (bf_t)f.y; o[2] = (bf_t)f.z; o[3] = (bf_t)f.w;
  dst[i] = o;
}

// dst[z][c][r] = (bf16) src[z][r][c]   (per-matrix transpose)
__global__ __launch_bounds__(256)
void cvt_transpose(const float* __restrict__ src, bf_t* __restrict__ dst,
                   int rows, int cols) {
  const int z = blockIdx.z;
  const size_t base = (size_t)z * rows * cols;
  const int idx = blockIdx.x * 256 + threadIdx.x;
  const int r = idx / cols, c = idx % cols;
  dst[base + (size_t)c * rows + r] = (bf_t)src[base + idx];
}

// ---------------------------------------------------------------------------
// C[M,N] = A[M,K] * Bt[N,K]^T ; bf16 in, bf16-out(*scale) or f32-out(+bias)
// Block: 256 thr (8 waves), tile 128x256, BK=32, async double-buffered LDS.
// Wave = 64x64 sub-tile (4x4 accums) -> 16 wmma : 16 ds_load_b128 per step.
// ---------------------------------------------------------------------------
template <bool OUT_BF16>
__global__ __launch_bounds__(256)
void gemm_bf16(const bf_t* __restrict__ A, const bf_t* __restrict__ Bt,
               void* __restrict__ Cout, const float* __restrict__ bias,
               int M, int N, int K, float scale) {
  __shared__ bf_t As[2][128 * 40];   // +8 pad keeps 16B alignment
  __shared__ bf_t Bs[2][256 * 40];
  const int t = threadIdx.x;
  const int lane = t & 31, wid = t >> 5;
  const int m0 = blockIdx.x * 128, n0 = blockIdx.y * 256;
  const int wm = (wid >> 2) * 64, wn = (wid & 3) * 64;
  const int ln = lane & 15, hi8 = (lane >> 4) * 8;
  const int arow = t >> 1, ahalf = (t & 1) * 16;

  v8f acc[4][4];
#pragma unroll
  for (int i = 0; i < 4; ++i)
#pragma unroll
    for (int j = 0; j < 4; ++j) acc[i][j] = {};

  auto stage = [&](int buf, int k0) {
    const bf_t* ga = A + (size_t)(m0 + arow) * K + k0 + ahalf;
    async_ld_b128(&As[buf][arow * 40 + ahalf],     ga);
    async_ld_b128(&As[buf][arow * 40 + ahalf + 8], ga + 8);
    const bf_t* gb = Bt + (size_t)(n0 + t) * K + k0;
#pragma unroll
    for (int j = 0; j < 4; ++j)
      async_ld_b128(&Bs[buf][t * 40 + j * 8], gb + j * 8);
  };

  stage(0, 0);
  wait_async0();
  __syncthreads();

  int cur = 0;
  for (int k0 = 0; k0 < K; k0 += 32) {
    if (k0 + 32 < K) stage(cur ^ 1, k0 + 32);  // overlap copy with WMMA
    v16bf bfr[4];
#pragma unroll
    for (int tn = 0; tn < 4; ++tn)
      bfr[tn] = ld_frag(&Bs[cur][(wn + tn * 16 + ln) * 40 + hi8]);
#pragma unroll
    for (int tm = 0; tm < 4; ++tm) {
      v16bf a = ld_frag(&As[cur][(wm + tm * 16 + ln) * 40 + hi8]);
#pragma unroll
      for (int tn = 0; tn < 4; ++tn)
        acc[tm][tn] = wmma_bf16(a, bfr[tn], acc[tm][tn]);
    }
    wait_async0();
    __syncthreads();
    cur ^= 1;
  }

  // C-frag: VGPR v -> row (cm8+v), lane%16 -> col (contiguous across lanes)
  const int cm8 = (lane >> 4) * 8;
#pragma unroll
  for (int tm = 0; tm < 4; ++tm) {
#pragma unroll
    for (int tn = 0; tn < 4; ++tn) {
      const int gm = m0 + wm + tm * 16 + cm8;
      const int gn = n0 + wn + tn * 16 + ln;
      if (OUT_BF16) {
        bf_t* C = (bf_t*)Cout;
#pragma unroll
        for (int v = 0; v < 8; ++v)
          C[(size_t)(gm + v) * N + gn] = (bf_t)(acc[tm][tn][v] * scale);
      } else {
        float* C = (float*)Cout;
        const float bv = bias[gn];
#pragma unroll
        for (int v = 0; v < 8; ++v)
          C[(size_t)(gm + v) * N + gn] = acc[tm][tn][v] + bv;
      }
    }
  }
}

// ---------------------------------------------------------------------------
// k_proj[b,h,r,d] = sum_n E[h,n,r] * K[b,n,h*64+d]  (and F/V -> v_projT)
// One block per (which,h,b): 256x64 output, K=4096. Et stored [h][r][n].
// which==0 -> kp stored [r][d] ; which==1 -> vpT stored [d][r].
// ---------------------------------------------------------------------------
__global__ __launch_bounds__(256)
void proj_kv(const bf_t* __restrict__ Et, const bf_t* __restrict__ Ft,
             const bf_t* __restrict__ Kb, const bf_t* __restrict__ Vb,
             bf_t* __restrict__ kp, bf_t* __restrict__ vpT) {
  __shared__ bf_t Es[256 * 40];
  __shared__ bf_t Bs[64 * 40];
  const int which = blockIdx.x, h = blockIdx.y, b = blockIdx.z;
  const bf_t* Eh = (which ? Ft : Et) + (size_t)h * RP * SEQ;
  const bf_t* S  = (which ? Vb : Kb);
  const int t = threadIdx.x, lane = t & 31, wid = t >> 5;
  const int wm = (wid >> 1) * 64, wn = (wid & 1) * 32;
  const int ln = lane & 15, hi8 = (lane >> 4) * 8;
  const int ni = t & 31, dj0 = (t >> 5) * 8;

  v8f acc[4][2];
#pragma unroll
  for (int i = 0; i < 4; ++i) { acc[i][0] = {}; acc[i][1] = {}; }

  for (int n0 = 0; n0 < SEQ; n0 += 32) {
    // A tile 256r x 32n, contiguous -> async copy
    const bf_t* ge = Eh + (size_t)t * SEQ + n0;
#pragma unroll
    for (int j = 0; j < 4; ++j)
      async_ld_b128(&Es[t * 40 + j * 8], ge + j * 8);
    // B tile (32n x 64d) -> LDS transposed [d][n] (VGPR-mediated)
    const bf_t* gs = S + (size_t)(b * SEQ + n0 + ni) * DM + h * DEPTH + dj0;
    v8bf sv = *(const v8bf*)gs;
#pragma unroll
    for (int jj = 0; jj < 8; ++jj) Bs[(dj0 + jj) * 40 + ni] = sv[jj];
    wait_async0();
    __syncthreads();

    v16bf b0 = ld_frag(Bs + (wn + ln) * 40 + hi8);
    v16bf b1 = ld_frag(Bs + (wn + 16 + ln) * 40 + hi8);
#pragma unroll
    for (int tm = 0; tm < 4; ++tm) {
      v16bf a = ld_frag(Es + (wm + tm * 16 + ln) * 40 + hi8);
      acc[tm][0] = wmma_bf16(a, b0, acc[tm][0]);
      acc[tm][1] = wmma_bf16(a, b1, acc[tm][1]);
    }
    __syncthreads();
  }

  const int cm8 = (lane >> 4) * 8;
  const size_t bh = (size_t)(b * NH + h);
#pragma unroll
  for (int tm = 0; tm < 4; ++tm) {
#pragma unroll
    for (int tn = 0; tn < 2; ++tn) {
      const int gm = wm + tm * 16 + cm8;   // r
      const int gn = wn + tn * 16 + ln;    // d
      if (which == 0) {
#pragma unroll
        for (int v = 0; v < 8; ++v)
          kp[bh * RP * DEPTH + (size_t)(gm + v) * DEPTH + gn] = (bf_t)acc[tm][tn][v];
      } else {
#pragma unroll
        for (int v = 0; v < 8; ++v)
          vpT[bh * DEPTH * RP + (size_t)gn * RP + gm + v] = (bf_t)acc[tm][tn][v];
      }
    }
  }
}

// ---------------------------------------------------------------------------
// Fused attention for one (b,h,64-token tile):
//   S = Qs(64x64) * kp^T(64x256) ; softmax over r ; O = A(64x256) * vp(256x64)
// Dynamic LDS: Qs 64x72 | KPs 256x72 | VPs 64x264 | Ss f32 64x260 | ATs 64x264
// Tiles staged by the Tensor Data Mover (one D# per tile, waves 0-2 issue).
// ---------------------------------------------------------------------------
__global__ __launch_bounds__(256)
void attn_softmax(const bf_t* __restrict__ Qb, const bf_t* __restrict__ kp,
                  const bf_t* __restrict__ vpT, bf_t* __restrict__ AO) {
  extern __shared__ __align__(16) char smem[];
  bf_t*  Qs  = (bf_t*)smem;               // 64 x 72
  bf_t*  KPs = Qs + 64 * 72;              // 256 x 72
  bf_t*  VPs = KPs + 256 * 72;            // 64 x 264
  float* Ss  = (float*)(VPs + 64 * 264);  // 64 x 260 fp32
  bf_t*  ATs = (bf_t*)(Ss + 64 * 260);    // 64 x 264

  const int nt = blockIdx.x, h = blockIdx.y, b = blockIdx.z;
  const int n0 = nt * 64;
  const size_t bh = (size_t)(b * NH + h);
  const int t = threadIdx.x, lane = t & 31, wid = t >> 5;
  const int ln = lane & 15, hi8 = (lane >> 4) * 8, cm8 = (lane >> 4) * 8;

#ifdef HAVE_TDM
  // One TDM descriptor per tile; TDM ignores EXEC, so gate by wave id only.
  // Q: 64 rows x 64 elems, row stride DM (head slice); pad 32dw rows by 4dw.
  if (wid == 0) {
    const bf_t* gq = Qb + (size_t)(b * SEQ + n0) * DM + h * DEPTH;
    tdm_load_2d(Qs, gq, DEPTH, 64, DM, /*interval 32dw*/4, /*pad 4dw*/3);
  } else if (wid == 1) {  // k_proj: 256 rows x 64 elems, contiguous
    tdm_load_2d(KPs, kp + bh * RP * DEPTH, DEPTH, RP, DEPTH, 4, 3);
  } else if (wid == 2) {  // v_projT: 64 rows x 256 elems; pad 128dw rows by 4dw
    tdm_load_2d(VPs, vpT + bh * DEPTH * RP, RP, DEPTH, RP, /*128dw*/6, 3);
  }
  __builtin_amdgcn_s_wait_tensorcnt(0);
#else
  {  // async fallback: stage Q tile (64 x 64)
    const int row = t >> 2, seg = (t & 3) * 16;
    const bf_t* gq = Qb + (size_t)(b * SEQ + n0 + row) * DM + h * DEPTH + seg;
    async_ld_b128(Qs + row * 72 + seg,     gq);
    async_ld_b128(Qs + row * 72 + seg + 8, gq + 8);
    const bf_t* gk = kp + bh * RP * DEPTH + (size_t)t * DEPTH;
#pragma unroll
    for (int j = 0; j < 8; ++j)
      async_ld_b128(KPs + t * 72 + j * 8, gk + j * 8);
    const int vrow = t >> 2, vseg = (t & 3) * 64;
    const bf_t* gv = vpT + bh * DEPTH * RP + (size_t)vrow * RP + vseg;
#pragma unroll
    for (int j = 0; j < 8; ++j)
      async_ld_b128(VPs + vrow * 264 + vseg + j * 8, gv + j * 8);
  }
  wait_async0();
#endif
  __syncthreads();

  {  // scores: each wave owns a 32-wide r strip of the 64x256 tile
    const int wn = wid * 32;
    v8f s[4][2];
#pragma unroll
    for (int i = 0; i < 4; ++i) { s[i][0] = {}; s[i][1] = {}; }
#pragma unroll
    for (int ks = 0; ks < 2; ++ks) {
      const int kb = ks * 32;
      v16bf b0 = ld_frag(KPs + (wn + ln) * 72 + kb + hi8);
      v16bf b1 = ld_frag(KPs + (wn + 16 + ln) * 72 + kb + hi8);
#pragma unroll
      for (int tm = 0; tm < 4; ++tm) {
        v16bf a = ld_frag(Qs + (tm * 16 + ln) * 72 + kb + hi8);
        s[tm][0] = wmma_bf16(a, b0, s[tm][0]);
        s[tm][1] = wmma_bf16(a, b1, s[tm][1]);
      }
    }
#pragma unroll
    for (int tm = 0; tm < 4; ++tm)
#pragma unroll
      for (int tn = 0; tn < 2; ++tn)
#pragma unroll
        for (int v = 0; v < 8; ++v)
          Ss[(tm * 16 + cm8 + v) * 260 + wn + tn * 16 + ln] = s[tm][tn][v];
  }
  __syncthreads();

  if (t < 64) {  // row softmax over r=256, write bf16 attn tile
    float* sr = Ss + t * 260;
    float mx = -3.4e38f;
    for (int c = 0; c < 256; c += 4) {
      float4 q = *(const float4*)(sr + c);
      mx = fmaxf(mx, fmaxf(fmaxf(q.x, q.y), fmaxf(q.z, q.w)));
    }
    float sum = 0.f;
    for (int c = 0; c < 256; c += 4) {
      float4 q = *(float4*)(sr + c);
      q.x = __expf(q.x - mx); q.y = __expf(q.y - mx);
      q.z = __expf(q.z - mx); q.w = __expf(q.w - mx);
      *(float4*)(sr + c) = q;
      sum += q.x + q.y + q.z + q.w;
    }
    const float inv = 1.0f / sum;
    bf_t* ar = ATs + t * 264;
    for (int c = 0; c < 256; c += 4) {
      float4 q = *(const float4*)(sr + c);
      *(unsigned*)(ar + c)     = pack_bf2(q.x * inv, q.y * inv);
      *(unsigned*)(ar + c + 2) = pack_bf2(q.z * inv, q.w * inv);
    }
  }
  __syncthreads();

  {  // O = attn(64x256) * vp(256x64): wave = 16x32 sub-tile, K=256
    const int wm = (wid >> 1) * 16, wn = (wid & 1) * 32;
    v8f o0 = {}, o1 = {};
#pragma unroll
    for (int ks = 0; ks < 8; ++ks) {
      const int kb = ks * 32;
      v16bf a  = ld_frag(ATs + (wm + ln) * 264 + kb + hi8);
      v16bf b0 = ld_frag(VPs + (wn + ln) * 264 + kb + hi8);
      v16bf b1 = ld_frag(VPs + (wn + 16 + ln) * 264 + kb + hi8);
      o0 = wmma_bf16(a, b0, o0);
      o1 = wmma_bf16(a, b1, o1);
    }
#pragma unroll
    for (int v = 0; v < 8; ++v) {
      const size_t gm = (size_t)(b * SEQ + n0 + wm + cm8 + v);
      AO[gm * DM + h * DEPTH + wn + ln]      = (bf_t)o0[v];
      AO[gm * DM + h * DEPTH + wn + 16 + ln] = (bf_t)o1[v];
    }
  }
}

// ---------------------------------------------------------------------------
extern "C" void kernel_launch(void* const* d_in, const int* in_sizes, int n_in,
                              void* d_out, int out_size, void* d_ws, size_t ws_size,
                              hipStream_t stream) {
  const float* x    = (const float*)d_in[0];
  const float* wq   = (const float*)d_in[1];
  const float* wk   = (const float*)d_in[2];
  const float* wv   = (const float*)d_in[3];
  const float* E    = (const float*)d_in[4];
  const float* F    = (const float*)d_in[5];
  const float* wout = (const float*)d_in[6];
  const float* bout = (const float*)d_in[7];

  bf_t* p = (bf_t*)d_ws;
  bf_t* xb  = p; p += (size_t)NTOK * DM;          // x bf16
  bf_t* wqT = p; p += (size_t)DM * DM;            // transposed weights
  bf_t* wkT = p; p += (size_t)DM * DM;
  bf_t* wvT = p; p += (size_t)DM * DM;
  bf_t* woT = p; p += (size_t)DM * DM;
  bf_t* EtB = p; p += (size_t)NH * RP * SEQ;      // E^T per head [r][n]
  bf_t* FtB = p; p += (size_t)NH * RP * SEQ;
  bf_t* Qb  = p; p += (size_t)NTOK * DM;
  bf_t* Kb  = p; p += (size_t)NTOK * DM;
  bf_t* Vb  = p; p += (size_t)NTOK * DM;
  bf_t* kpB = p; p += (size_t)NB * NH * RP * DEPTH;  // [bh][r][d]
  bf_t* vpB = p; p += (size_t)NB * NH * DEPTH * RP;  // [bh][d][r]
  bf_t* AO  = Kb;  // K consumed by proj_kv before attn writes AO (aliased)

  // fp32 -> bf16 converts (weights/E/F transposed for [N,K] fragment loads)
  cvt_copy4<<<(NTOK * DM / 4) / 256, 256, 0, stream>>>((const float4*)x, (v4bf*)xb);
  cvt_transpose<<<dim3((DM * DM) / 256, 1, 1), 256, 0, stream>>>(wq, wqT, DM, DM);
  cvt_transpose<<<dim3((DM * DM) / 256, 1, 1), 256, 0, stream>>>(wk, wkT, DM, DM);
  cvt_transpose<<<dim3((DM * DM) / 256, 1, 1), 256, 0, stream>>>(wv, wvT, DM, DM);
  cvt_transpose<<<dim3((DM * DM) / 256, 1, 1), 256, 0, stream>>>(wout, woT, DM, DM);
  cvt_transpose<<<dim3((SEQ * RP) / 256, 1, NH), 256, 0, stream>>>(E, EtB, SEQ, RP);
  cvt_transpose<<<dim3((SEQ * RP) / 256, 1, NH), 256, 0, stream>>>(F, FtB, SEQ, RP);

  // QKV projections (Q pre-scaled by 1/sqrt(DEPTH) = 0.125)
  dim3 gg(NTOK / 128, DM / 256);
  gemm_bf16<true><<<gg, 256, 0, stream>>>(xb, wqT, Qb, nullptr, NTOK, DM, DM, 0.125f);
  gemm_bf16<true><<<gg, 256, 0, stream>>>(xb, wkT, Kb, nullptr, NTOK, DM, DM, 1.0f);
  gemm_bf16<true><<<gg, 256, 0, stream>>>(xb, wvT, Vb, nullptr, NTOK, DM, DM, 1.0f);

  // Linformer sequence projection of K and V
  proj_kv<<<dim3(2, NH, NB), 256, 0, stream>>>(EtB, FtB, Kb, Vb, kpB, vpB);

  // Fused scores + softmax + attn*V (176 KB dynamic LDS, <320 KB/WGP)
  const size_t smem = (size_t)(64 * 72 + 256 * 72 + 64 * 264) * sizeof(bf_t) +
                      (size_t)(64 * 260) * sizeof(float) +
                      (size_t)(64 * 264) * sizeof(bf_t);
  (void)hipFuncSetAttribute(reinterpret_cast<const void*>(attn_softmax),
                            hipFuncAttributeMaxDynamicSharedMemorySize, (int)smem);
  attn_softmax<<<dim3(SEQ / 64, NH, NB), 256, smem, stream>>>(Qb, kpB, vpB, AO);

  // Output projection + bias -> fp32 d_out
  gemm_bf16<false><<<gg, 256, 0, stream>>>(AO, woT, d_out, bout, NTOK, DM, DM, 1.0f);
}
